// WaveGNN_37074157699472
// MI455X (gfx1250) — matrix operand order repaired and checked by reading, not compiled, not run on hardware
//
#include <hip/hip_runtime.h>

#define B_ 16
#define N_ 2048
#define D_ 128
#define L_ 3

typedef __attribute__((ext_vector_type(16))) __bf16 bf16x16;
typedef __attribute__((ext_vector_type(2)))  __bf16 bf16x2;
typedef __attribute__((ext_vector_type(8)))  float  f32x8;
typedef __attribute__((ext_vector_type(4)))  float  f32x4;

// Fallback path only (no adjT workspace): batches [0, ADJ_RT_BATCHES) of adj
// keep regular temporal hints (stay L2-resident across layers), the rest
// streams NT so it cannot evict the resident portion.
#define ADJ_RT_BATCHES 9

__device__ __forceinline__ unsigned short f2bf(float x) {
  // round-to-nearest-even fp32 -> bf16
  unsigned int u = __float_as_uint(x);
  unsigned int r = (u + 0x7FFFu + ((u >> 16) & 1u)) >> 16;
  return (unsigned short)r;
}

// Convert+pack two fp32 -> one dword {hi:bf16(b), lo:bf16(a)}.
__device__ __forceinline__ unsigned int pack2_bf16(float a, float b) {
#if __has_builtin(__builtin_amdgcn_cvt_pk_bf16_f32)
  union { bf16x2 v; unsigned int u; } cv;
  cv.v = __builtin_amdgcn_cvt_pk_bf16_f32(a, b);
  return cv.u;
#else
  // round-to-nearest (ties away)
  const unsigned int lo = (__float_as_uint(a) + 0x8000u) >> 16;
  const unsigned int hi = (__float_as_uint(b) + 0x8000u) & 0xFFFF0000u;
  return hi | lo;
#endif
}

// ---------------------------------------------------------------------------
// Kernel 0 (run once, preferred path): adjT[b][j][i] = bf16(adj[b][i][j]).
// 64x64 LDS tile transpose. adj is streamed NT (read exactly once); the
// 128 MiB bf16 result fits in the 192 MB L2 and is re-read by all 3 layers
// at L2 bandwidth.
// ---------------------------------------------------------------------------
__global__ __launch_bounds__(256) void adjT_convert_kernel(
    const float* __restrict__ adj, unsigned short* __restrict__ adjT) {
  __shared__ __align__(16) float tileT[64][68];   // [j_local][i_local], padded
  const int b   = blockIdx.z;
  const int i0  = blockIdx.x * 64;
  const int j0  = blockIdx.y * 64;
  const int tid = threadIdx.x;
  const int rl  = tid >> 2;          // 0..63
  const int cg  = (tid & 3) * 16;    // 0,16,32,48

  // Phase 1: read 16 floats of adj row i=i0+rl, cols j0+cg.. ; store transposed
  const f32x4* src =
      (const f32x4*)(adj + (size_t)b * N_ * N_ + (size_t)(i0 + rl) * N_ + j0 + cg);
#pragma unroll
  for (int q = 0; q < 4; ++q) {
    const f32x4 v = __builtin_nontemporal_load(src + q);
    tileT[cg + q * 4 + 0][rl] = v[0];
    tileT[cg + q * 4 + 1][rl] = v[1];
    tileT[cg + q * 4 + 2][rl] = v[2];
    tileT[cg + q * 4 + 3][rl] = v[3];
  }
  __syncthreads();

  // Phase 2: write adjT row j=j0+rl, cols i0+cg.. (16 bf16 = 32B, 2x b128)
  const float* trow = &tileT[rl][cg];
  unsigned int o0, o1, o2, o3, o4, o5, o6, o7;
  {
    const f32x4 v0 = *(const f32x4*)(trow + 0);
    const f32x4 v1 = *(const f32x4*)(trow + 4);
    const f32x4 v2 = *(const f32x4*)(trow + 8);
    const f32x4 v3 = *(const f32x4*)(trow + 12);
    o0 = pack2_bf16(v0[0], v0[1]);  o1 = pack2_bf16(v0[2], v0[3]);
    o2 = pack2_bf16(v1[0], v1[1]);  o3 = pack2_bf16(v1[2], v1[3]);
    o4 = pack2_bf16(v2[0], v2[1]);  o5 = pack2_bf16(v2[2], v2[3]);
    o6 = pack2_bf16(v3[0], v3[1]);  o7 = pack2_bf16(v3[2], v3[3]);
  }
  uint4* dst =
      (uint4*)(adjT + (size_t)b * N_ * N_ + (size_t)(j0 + rl) * N_ + i0 + cg);
  dst[0] = make_uint4(o0, o1, o2, o3);
  dst[1] = make_uint4(o4, o5, o6, o7);
}

// ---------------------------------------------------------------------------
// Kernel 1: hwT[b][e][i] = bf16( sum_d h[b][i][d] * W[d][e] )
// ---------------------------------------------------------------------------
__global__ __launch_bounds__(128) void lin_bf16t_kernel(
    const float* __restrict__ h, const float* __restrict__ W,
    unsigned short* __restrict__ hwT) {
  __shared__ float row[D_];
  const int t   = threadIdx.x;
  const int bid = blockIdx.x;          // b*N + i
  const int b   = bid >> 11;
  const int i   = bid & (N_ - 1);
  row[t] = h[(size_t)bid * D_ + t];
  __syncthreads();
  float acc = 0.f;
#pragma unroll 8
  for (int d = 0; d < D_; ++d) acc = fmaf(row[d], W[d * D_ + t], acc);
  hwT[((size_t)b * D_ + t) * N_ + i] = f2bf(acc);
}

// ---------------------------------------------------------------------------
// Staged tile data, kept strictly in registers (plain struct returned by
// value from a free function: no allocas, no lambda reference captures, so
// SROA succeeds and AMDGPUPromoteAlloca cannot demote it to LDS).
// ---------------------------------------------------------------------------
struct Stage {
  uint4 a0, a1;   // A tile: 16 bf16 (PRE path) or 8 packed k-pair dwords
  uint4 b0, b1;   // B tile: 16 bf16
};

template <bool PRE_BF16>
static __device__ __forceinline__ Stage stage_load(
    const unsigned short* __restrict__ adjTB, const float* __restrict__ adjB,
    const unsigned short* __restrict__ hwB, int j0, int k0, int s_row,
    int s_hs, int a_kp, int a_jg, bool adjNT) {
  Stage s;
  if constexpr (PRE_BF16) {
    const uint4* sa =
        (const uint4*)(adjTB + (size_t)(j0 + s_row) * N_ + k0 + s_hs);
    s.a0 = sa[0];
    s.a1 = sa[1];
  } else {
    const f32x4* p0 =
        (const f32x4*)(adjB + (size_t)(k0 + 2 * a_kp) * N_ + j0 + a_jg);
    const f32x4* p1 = (const f32x4*)((const float*)p0 + N_);
    f32x4 r00, r01, r10, r11;
    if (adjNT) {
      r00 = __builtin_nontemporal_load(p0);
      r01 = __builtin_nontemporal_load(p0 + 1);
      r10 = __builtin_nontemporal_load(p1);
      r11 = __builtin_nontemporal_load(p1 + 1);
    } else {
      r00 = p0[0];
      r01 = p0[1];
      r10 = p1[0];
      r11 = p1[1];
    }
    // pack k-pairs now: dword u = {lo: bf16(row k even, j=u), hi: k odd}
    s.a0.x = pack2_bf16(r00[0], r10[0]);
    s.a0.y = pack2_bf16(r00[1], r10[1]);
    s.a0.z = pack2_bf16(r00[2], r10[2]);
    s.a0.w = pack2_bf16(r00[3], r10[3]);
    s.a1.x = pack2_bf16(r01[0], r11[0]);
    s.a1.y = pack2_bf16(r01[1], r11[1]);
    s.a1.z = pack2_bf16(r01[2], r11[2]);
    s.a1.w = pack2_bf16(r01[3], r11[3]);
  }
  const uint4* sb = (const uint4*)(hwB + (size_t)s_row * N_ + k0 + s_hs);
  s.b0 = sb[0];
  s.b1 = sb[1];
  return s;
}

// ---------------------------------------------------------------------------
// Kernel 2: agg = adj^T @ hw (bf16 WMMA, f32 acc), z = agg + bias + residual.
// Block = (b, 128-row j-tile) x 128 cols; 8 waves, wave (r,c) owns 32x64 =
// 2x4 WMMA tiles. Software-pipelined register prefetch of the next tile.
// ---------------------------------------------------------------------------
template <bool PRE_BF16>
__global__ __launch_bounds__(256, 2) void gcn_gemm_kernel(
    const float* __restrict__ adjF,          // [B][N][N] fp32 (fallback path)
    const unsigned short* __restrict__ adjT, // [B][N][N] bf16 transposed
    const unsigned short* __restrict__ hwT,  // [B][D][N] bf16, K-contiguous
    const float* __restrict__ hin,           // [B][N][D] residual
    const float* __restrict__ bias,          // [D]
    float* __restrict__ zbuf)                // [B][N][D]
{
  __shared__ __align__(16) unsigned short As[128][40]; // [j][k], padded stride
  __shared__ __align__(16) unsigned short Bs[128][40]; // [e][k], padded stride

  const int b    = blockIdx.y;
  const int j0   = blockIdx.x * 128;
  const int tid  = threadIdx.x;
  const int lane = tid & 31;
  const int wave = tid >> 5;
  const int r    = wave & 3;   // row sub-block (32 rows)
  const int c    = wave >> 2;  // col sub-block (64 cols)
  const int ln16 = lane & 15;
  const int sel  = lane >> 4;

  f32x8 acc[2][4] = {};

  const unsigned short* hwB = hwT + (size_t)b * D_ * N_;

  // Row/half staging map (used by B always, and by A when PRE_BF16)
  const int s_row = tid >> 1;         // 0..127
  const int s_hs  = (tid & 1) * 16;   // half offset (16 bf16 = 32B)
  // fp32 A staging map: thread owns k-pair a_kp (2 adj rows) x 8 consecutive j
  const int a_kp = tid >> 4;          // 0..15
  const int a_jg = (tid & 15) * 8;    // j group of 8
  const bool adjNT = (b >= ADJ_RT_BATCHES);

  const float*          adjB  = adjF + (size_t)b * N_ * N_;
  const unsigned short* adjTB = adjT + (size_t)b * N_ * N_;

  Stage st = stage_load<PRE_BF16>(adjTB, adjB, hwB, j0, 0, s_row, s_hs,
                                  a_kp, a_jg, adjNT);

  for (int k0 = 0; k0 < N_; k0 += 32) {
    // --- commit staged registers to LDS ---
    if constexpr (PRE_BF16) {
      uint4* d4 = (uint4*)&As[s_row][s_hs];
      d4[0] = st.a0;
      d4[1] = st.a1;
    } else {
      *(unsigned int*)&As[a_jg + 0][2 * a_kp] = st.a0.x;
      *(unsigned int*)&As[a_jg + 1][2 * a_kp] = st.a0.y;
      *(unsigned int*)&As[a_jg + 2][2 * a_kp] = st.a0.z;
      *(unsigned int*)&As[a_jg + 3][2 * a_kp] = st.a0.w;
      *(unsigned int*)&As[a_jg + 4][2 * a_kp] = st.a1.x;
      *(unsigned int*)&As[a_jg + 5][2 * a_kp] = st.a1.y;
      *(unsigned int*)&As[a_jg + 6][2 * a_kp] = st.a1.z;
      *(unsigned int*)&As[a_jg + 7][2 * a_kp] = st.a1.w;
    }
    {
      uint4* d4 = (uint4*)&Bs[s_row][s_hs];
      d4[0] = st.b0;
      d4[1] = st.b1;
    }
    __syncthreads();

    // --- fragment loads (ISA bf16 16x32 A / 32x16 B VGPR layouts) ---
    union Frag { uint4 u[2]; bf16x16 v; };
    Frag af[2], bfr[4];
#pragma unroll
    for (int mi = 0; mi < 2; ++mi) {
      const unsigned short* p = &As[r * 32 + mi * 16 + ln16][sel * 8];
      af[mi].u[0] = *(const uint4*)p;         // VGPR0-3: K=sel*8 + 0..7
      af[mi].u[1] = *(const uint4*)(p + 16);  // VGPR4-7: K=sel*8 + 16..23
    }
#pragma unroll
    for (int ni = 0; ni < 4; ++ni) {
      const unsigned short* p = &Bs[c * 64 + ni * 16 + ln16][sel * 8];
      bfr[ni].u[0] = *(const uint4*)p;         // K = sel*8 + 0..7
      bfr[ni].u[1] = *(const uint4*)(p + 16);  // K = sel*8 + 16..23
    }

    // --- prefetch next tile's global data (overlaps with WMMA below) ---
    if (k0 + 32 < N_)
      st = stage_load<PRE_BF16>(adjTB, adjB, hwB, j0, k0 + 32, s_row, s_hs,
                                a_kp, a_jg, adjNT);

#pragma unroll
    for (int mi = 0; mi < 2; ++mi)
#pragma unroll
      for (int ni = 0; ni < 4; ++ni)
        acc[mi][ni] = __builtin_amdgcn_wmma_f32_16x16x32_bf16(
            false, af[mi].v, false, bfr[ni].v, (short)0, acc[mi][ni],
            false, false);
    __syncthreads();   // all waves done reading LDS before next commit
  }

  // --- epilogue: z = agg + bias + residual ---
#pragma unroll
  for (int mi = 0; mi < 2; ++mi)
#pragma unroll
    for (int ni = 0; ni < 4; ++ni) {
      const int e  = c * 64 + ni * 16 + ln16;
      const int jb = j0 + r * 32 + mi * 16 + sel * 8;
      const float bia = bias[e];
#pragma unroll
      for (int g = 0; g < 8; ++g) {
        const size_t idx = ((size_t)b * N_ + jb + g) * D_ + e;
        zbuf[idx] = acc[mi][ni][g] + bia + hin[idx];
      }
    }
}

// ---------------------------------------------------------------------------
// Kernel 3: per-row LayerNorm(D=128) + ReLU (wave32 shuffle reductions).
// ---------------------------------------------------------------------------
__global__ __launch_bounds__(256) void ln_relu_kernel(
    const float* __restrict__ zbuf, const float* __restrict__ gamma,
    const float* __restrict__ beta, float* __restrict__ out) {
  const int lane = threadIdx.x & 31;
  const int wave = threadIdx.x >> 5;
  const size_t row  = (size_t)blockIdx.x * 8 + wave;
  const size_t base = row * D_ + lane * 4;
  float4 z = *(const float4*)(zbuf + base);
  float s = z.x + z.y + z.z + z.w;
#pragma unroll
  for (int off = 16; off > 0; off >>= 1) s += __shfl_xor(s, off, 32);
  const float mean = s * (1.f / 128.f);
  const float dx = z.x - mean, dy = z.y - mean, dz = z.z - mean, dw = z.w - mean;
  float q = dx * dx + dy * dy + dz * dz + dw * dw;
#pragma unroll
  for (int off = 16; off > 0; off >>= 1) q += __shfl_xor(q, off, 32);
  const float inv = rsqrtf(q * (1.f / 128.f) + 1e-5f);
  const float4 g4  = *(const float4*)(gamma + lane * 4);
  const float4 be4 = *(const float4*)(beta + lane * 4);
  float4 o;
  o.x = fmaxf(0.f, dx * inv * g4.x + be4.x);
  o.y = fmaxf(0.f, dy * inv * g4.y + be4.y);
  o.z = fmaxf(0.f, dz * inv * g4.z + be4.z);
  o.w = fmaxf(0.f, dw * inv * g4.w + be4.w);
  *(float4*)(out + base) = o;
}

// ---------------------------------------------------------------------------
extern "C" void kernel_launch(void* const* d_in, const int* in_sizes, int n_in,
                              void* d_out, int out_size, void* d_ws, size_t ws_size,
                              hipStream_t stream) {
  (void)in_sizes; (void)n_in; (void)out_size;
  const float* X   = (const float*)d_in[0];  // [B][N][D]
  const float* adj = (const float*)d_in[1];  // [B][N][N]
  const float* Ws  = (const float*)d_in[2];  // [L][D][D]
  const float* bs  = (const float*)d_in[3];  // [L][D]
  const float* gs  = (const float*)d_in[4];  // [L][D]
  const float* bes = (const float*)d_in[5];  // [L][D]
  float* out = (float*)d_out;
  char*  ws  = (char*)d_ws;

  unsigned short* hwT = (unsigned short*)ws;             // 8 MiB  (B*D*N*2)
  float* zbuf = (float*)(ws + (8u  << 20));              // 16 MiB
  float* hbuf = (float*)(ws + (24u << 20));              // 16 MiB
  unsigned short* adjT = (unsigned short*)(ws + (40u << 20));   // 128 MiB
  const size_t need = (size_t)(40u << 20) + (size_t)B_ * N_ * N_ * 2;
  const bool pre = (ws_size >= need);   // ws_size is fixed -> deterministic

  if (pre)
    adjT_convert_kernel<<<dim3(N_ / 64, N_ / 64, B_), 256, 0, stream>>>(adj, adjT);

  const dim3 gGemm(N_ / 128, B_);
  for (int l = 0; l < L_; ++l) {
    const float* hin  = (l == 0)      ? X   : hbuf;
    float*       hout = (l == L_ - 1) ? out : hbuf;
    lin_bf16t_kernel<<<B_ * N_, 128, 0, stream>>>(hin, Ws + (size_t)l * D_ * D_, hwT);
    if (pre)
      gcn_gemm_kernel<true><<<gGemm, 256, 0, stream>>>(adj, adjT, hwT, hin,
                                                       bs + l * D_, zbuf);
    else
      gcn_gemm_kernel<false><<<gGemm, 256, 0, stream>>>(adj, adjT, hwT, hin,
                                                        bs + l * D_, zbuf);
    ln_relu_kernel<<<(B_ * N_) / 8, 256, 0, stream>>>(zbuf, gs + l * D_,
                                                      bes + l * D_, hout);
  }
}
